// MultiheadDilatedAttention_76673756168936
// MI455X (gfx1250) — matrix operand order repaired and checked by reading, not compiled, not run on hardware
//
#include <hip/hip_runtime.h>
#include <cstdint>
#include <cstddef>

typedef __attribute__((ext_vector_type(16))) __bf16 v16bf;
typedef __attribute__((ext_vector_type(8)))  __bf16 v8bf;
typedef __attribute__((ext_vector_type(4)))  __bf16 v4bf;
typedef __attribute__((ext_vector_type(8)))  float  v8f;

#define E_DIM 768
#define SEQ_N 8192
#define NHEAD 12
#define HDIM  64

// float -> bf16, round-to-nearest-even
static __device__ inline __bf16 f2bf(float f) {
  union { float f; uint32_t u; } in; in.f = f;
  uint32_t u = in.u;
  uint32_t r = u + 0x7FFFu + ((u >> 16) & 1u);
  union { uint16_t u; __bf16 b; } out; out.u = (uint16_t)(r >> 16);
  return out.b;
}

// build a v16bf fragment from two 16-byte-aligned 8-element chunks
static __device__ inline v16bf ld8x2(const __bf16* p0, const __bf16* p1) {
  v8bf lo = *(const v8bf*)p0;
  v8bf hi = *(const v8bf*)p1;
  v16bf r;
#pragma unroll
  for (int i = 0; i < 8; ++i) { r[i] = lo[i]; r[i + 8] = hi[i]; }
  return r;
}

static __device__ inline v8f wmma_bf16(v16bf a, v16bf b, v8f c) {
  // D(16x16,f32) = A(16x32,bf16) * B(32x16,bf16) + C
  return __builtin_amdgcn_wmma_f32_16x16x32_bf16(false, a, false, b, (short)0, c,
                                                 false, false);
}

// ---- VALU-only 16-lane xor reductions via DPP ROW_XMASK (0x160|mask) ------
template <int MASK>
static __device__ inline float row_xor_max(float v) {
  int o = __builtin_amdgcn_update_dpp(__float_as_int(v), __float_as_int(v),
                                      0x160 | MASK, 0xF, 0xF, false);
  return fmaxf(v, __int_as_float(o));
}
template <int MASK>
static __device__ inline float row_xor_add(float v) {
  int o = __builtin_amdgcn_update_dpp(__float_as_int(v), __float_as_int(v),
                                      0x160 | MASK, 0xF, 0xF, false);
  return v + __int_as_float(o);
}
static __device__ inline float red16_max(float v) {
  v = row_xor_max<1>(v); v = row_xor_max<2>(v);
  v = row_xor_max<4>(v); v = row_xor_max<8>(v);
  return v;
}
static __device__ inline float red16_add(float v) {
  v = row_xor_add<1>(v); v = row_xor_add<2>(v);
  v = row_xor_add<4>(v); v = row_xor_add<8>(v);
  return v;
}

// low 32 bits of a flat LDS pointer == wave-relative LDS byte address
static __device__ inline unsigned lds_off(const void* p) {
  return (unsigned)(uintptr_t)p;
}

// ---------------- staging helpers (overload on input dtype) ----------------
static __device__ inline void stage4(const float* p, __bf16* dst) {
  float4 av = *(const float4*)p;
  dst[0] = f2bf(av.x); dst[1] = f2bf(av.y);
  dst[2] = f2bf(av.z); dst[3] = f2bf(av.w);
}
static __device__ inline void stage4(const __bf16* p, __bf16* dst) {
  v4bf av = *(const v4bf*)p;
#pragma unroll
  for (int i = 0; i < 4; ++i) dst[i] = av[i];
}

// ---------------------------------------------------------------------------
// C[M x 768] = A[M x 768] @ W[768 x 768] + bias    (BM=64, BN=128, BK=32)
// 8 waves in 2x4; each wave owns a 32x32 macro-tile = 2x2 WMMA tiles,
// so 4 WMMAs per 4 fragment loads per k-step.
// ---------------------------------------------------------------------------
template <typename TIN, bool OUT_BF16>
__global__ __launch_bounds__(256) void gemm768_kernel(
    const TIN* __restrict__ A, const float* __restrict__ W,
    const float* __restrict__ bias, void* __restrict__ out, int M)
{
  __shared__ __bf16 sA[64][40];    // [m][k] bf16, rows 16B-aligned (80B pitch)
  __shared__ __bf16 sB[128][40];   // [n][k] bf16 (transposed W tile)

  const int tid  = threadIdx.x;
  const int bm   = blockIdx.x * 64;
  const int bn   = blockIdx.y * 128;
  const int wave = tid >> 5, lane = tid & 31;
  const int hl   = lane >> 4, ln = lane & 15;
  const int wm   = wave >> 2, wn = wave & 3;

  v8f acc[2][2];
#pragma unroll
  for (int mt = 0; mt < 2; ++mt)
#pragma unroll
    for (int nt = 0; nt < 2; ++nt)
#pragma unroll
      for (int j = 0; j < 8; ++j) acc[mt][nt][j] = 0.0f;

  const int arow = tid >> 2, ac8 = (tid & 3) * 8;    // A: 64 rows x 8 elems
  const int bk   = tid >> 3, bn16 = (tid & 7) * 16;  // W: 32 k-rows x 16 n

  for (int k0 = 0; k0 < E_DIM; k0 += 32) {
    __syncthreads();
    // stage A (64x32) as bf16
    const TIN* ap = A + (size_t)(bm + arow) * E_DIM + k0 + ac8;
    stage4(ap,     &sA[arow][ac8]);
    stage4(ap + 4, &sA[arow][ac8 + 4]);
    // stage W tile (32x128) transposed -> [n][k]
    {
      const float* wp = W + (size_t)(k0 + bk) * E_DIM + bn + bn16;
#pragma unroll
      for (int c = 0; c < 4; ++c) {
        float4 w = *(const float4*)(wp + c * 4);
        sB[bn16 + c * 4 + 0][bk] = f2bf(w.x);
        sB[bn16 + c * 4 + 1][bk] = f2bf(w.y);
        sB[bn16 + c * 4 + 2][bk] = f2bf(w.z);
        sB[bn16 + c * 4 + 3][bk] = f2bf(w.w);
      }
    }
    if (k0 + 32 < E_DIM) {   // prefetch next k-tile (global_prefetch path)
      __builtin_prefetch(ap + 32, 0, 0);
      __builtin_prefetch(W + (size_t)(k0 + 32 + bk) * E_DIM + bn + bn16, 0, 0);
    }
    __syncthreads();

    v16bf afrag[2], bfrag[2];
#pragma unroll
    for (int mt = 0; mt < 2; ++mt) {
      const __bf16* p = &sA[wm * 32 + mt * 16 + ln][hl * 8];
      afrag[mt] = ld8x2(p, p + 16);      // A: K = hl*8+{0..7}, hl*8+16+{0..7}
    }
#pragma unroll
    for (int nt = 0; nt < 2; ++nt) {
      const __bf16* p = &sB[wn * 32 + nt * 16 + ln][hl * 8];
      bfrag[nt] = ld8x2(p, p + 16);      // B: K = hl*8+{0..7}, hl*8+16+{0..7}
    }
#pragma unroll
    for (int mt = 0; mt < 2; ++mt)
#pragma unroll
      for (int nt = 0; nt < 2; ++nt)
        acc[mt][nt] = wmma_bf16(afrag[mt], bfrag[nt], acc[mt][nt]);
  }

#pragma unroll
  for (int nt = 0; nt < 2; ++nt) {
    const int ncol = bn + wn * 32 + nt * 16 + ln;
    const float bv = bias[ncol];
#pragma unroll
    for (int mt = 0; mt < 2; ++mt)
#pragma unroll
      for (int j = 0; j < 8; ++j) {      // D: VGPR j <-> M = j + 8*hl
        int m = bm + wm * 32 + mt * 16 + hl * 8 + j;
        float v = acc[mt][nt][j] + bv;
        if (OUT_BF16) ((__bf16*)out)[(size_t)m * E_DIM + ncol] = f2bf(v);
        else          ((float*)out)[(size_t)m * E_DIM + ncol] = v;
      }
  }
}

// ---------------------------------------------------------------------------
// Dilated flash attention. Block = 4 waves = 64 query rows of one
// (group, batch, segment, head). Streams 32 kv positions per chunk.
// K chunk staged with GLOBAL_LOAD_ASYNC_TO_LDS (ASYNCcnt); V staged
// transposed through VGPRs. Softmax reductions are VALU DPP only.
// ---------------------------------------------------------------------------
__global__ __launch_bounds__(128) void dilated_attn_kernel(
    const __bf16* __restrict__ Q, const __bf16* __restrict__ K,
    const __bf16* __restrict__ V, float* __restrict__ att)
{
  __shared__ __bf16 sK[32][72];       // [kv][feat]
  __shared__ __bf16 sV[64][40];       // [feat][kv]  (transposed)
  __shared__ __bf16 sP[4][16][32];    // per-wave P tile, D->A layout change

  int gid = blockIdx.x;
  int g, t;
  if (gid < 1024)      { g = 0; t = gid; }
  else if (gid < 1536) { g = 1; t = gid - 1024; }
  else                 { g = 2; t = gid - 1536; }
  const int dil[3]  = {1, 2, 4};
  const int seg[3]  = {2048, 4096, 8192};
  const int nsg[3]  = {4, 2, 1};
  const int offc[3] = {0, 1, 2};
  const int r_ = dil[g], s_ = seg[g], off = offc[g];
  int qblk = t & 31; t >>= 5;
  int hg   = t & 3;  t >>= 2;
  int n    = t % nsg[g];
  int bi   = t / nsg[g];
  int head = 4 * g + hg;

  const int tid  = threadIdx.x;
  const int wave = tid >> 5, lane = tid & 31;
  const int hl   = lane >> 4, ln = lane & 15;

  // Q fragments (lane row = l&15; registers for both 32-wide k-steps)
  int jq = qblk * 64 + wave * 16 + ln;
  size_t qrow = ((size_t)(bi * SEQ_N + (n * s_ + off + r_ * jq))) * E_DIM
                + head * HDIM;
  v16bf qf[2];
#pragma unroll
  for (int ks = 0; ks < 2; ++ks) {
    const __bf16* p = Q + qrow + ks * 32 + hl * 8;
    qf[ks] = ld8x2(p, p + 16);
  }

  v8f o[4];
  float mrun[8], lrun[8];
#pragma unroll
  for (int nt = 0; nt < 4; ++nt)
#pragma unroll
    for (int j = 0; j < 8; ++j) o[nt][j] = 0.0f;
#pragma unroll
  for (int j = 0; j < 8; ++j) { mrun[j] = -3.0e38f; lrun[j] = 0.0f; }

  const int kvr = tid >> 2;           // 0..31: kv row staged by this thread
  const int f0  = (tid & 3) * 16;     // 16 features per thread

  const unsigned ldsK = lds_off(&sK[kvr][f0]);

  for (int c = 0; c < 64; ++c) {
    __syncthreads();
    // ---- K chunk: async DMA straight into LDS (no VGPR roundtrip) ----
    int posk = n * s_ + off + r_ * (c * 32 + kvr);
    size_t krow = ((size_t)(bi * SEQ_N + posk)) * E_DIM + head * HDIM + f0;
    {
      unsigned long long ka0 = (unsigned long long)(uintptr_t)(K + krow);
      unsigned long long ka1 = (unsigned long long)(uintptr_t)(K + krow + 8);
      asm volatile("global_load_async_to_lds_b128 %0, %1, off"
                   :: "v"(ldsK), "v"(ka0) : "memory");
      asm volatile("global_load_async_to_lds_b128 %0, %1, off"
                   :: "v"(ldsK + 16u), "v"(ka1) : "memory");
    }
    // ---- V chunk: through VGPRs, transposed into [feat][kv] ----
    v8bf v0 = *(const v8bf*)(V + krow);
    v8bf v1 = *(const v8bf*)(V + krow + 8);
#pragma unroll
    for (int i = 0; i < 8; ++i) {
      sV[f0 + i][kvr]     = v0[i];
      sV[f0 + 8 + i][kvr] = v1[i];
    }
    asm volatile("s_wait_asynccnt 0" ::: "memory");
    __syncthreads();

    // ---- S = Q K^T for two 16-key tiles (K-dim 64 = 2 WMMAs each) ----
    v8f sc[2];
#pragma unroll
    for (int kt = 0; kt < 2; ++kt) {
      v8f a;
#pragma unroll
      for (int j = 0; j < 8; ++j) a[j] = 0.0f;
      const __bf16* kp = &sK[kt * 16 + ln][hl * 8];
      a = wmma_bf16(qf[0], ld8x2(kp,      kp + 16), a);
      a = wmma_bf16(qf[1], ld8x2(kp + 32, kp + 48), a);
      sc[kt] = a;
    }

    // ---- online softmax; slot j: all 16 lanes of a half hold row j+8*hl
    float corr[8];
#pragma unroll
    for (int j = 0; j < 8; ++j) {
      float s0 = sc[0][j] * 0.125f;           // 1/sqrt(64)
      float s1 = sc[1][j] * 0.125f;
      float tmax = red16_max(fmaxf(s0, s1));  // DPP, VALU-only
      float mnew = fmaxf(mrun[j], tmax);
      corr[j] = __expf(mrun[j] - mnew);
      mrun[j] = mnew;
      float p0 = __expf(s0 - mnew);
      float p1 = __expf(s1 - mnew);
      float rs = red16_add(p0 + p1);          // DPP, VALU-only
      lrun[j] = lrun[j] * corr[j] + rs;
      int m = hl * 8 + j;                     // D layout -> P row
      sP[wave][m][ln]      = f2bf(p0);
      sP[wave][m][16 + ln] = f2bf(p1);
    }
#pragma unroll
    for (int nt = 0; nt < 4; ++nt)
#pragma unroll
      for (int j = 0; j < 8; ++j) o[nt][j] *= corr[j];

    __syncthreads();                          // P tile visible (A-layout read)

    // ---- O += P V : A frag from sP, B frags contiguous from sV ----
    const __bf16* pp = &sP[wave][ln][hl * 8];
    v16bf pfrag = ld8x2(pp, pp + 16);
#pragma unroll
    for (int nt = 0; nt < 4; ++nt) {
      const __bf16* vp = &sV[nt * 16 + ln][hl * 8];
      o[nt] = wmma_bf16(pfrag, ld8x2(vp, vp + 16), o[nt]);
    }
  }

  // ---- epilogue: O / l, scatter to att (disjoint pos/head across blocks)
#pragma unroll
  for (int j = 0; j < 8; ++j) {
    int m = hl * 8 + j;
    int jq2 = qblk * 64 + wave * 16 + m;
    size_t orow = ((size_t)(bi * SEQ_N + (n * s_ + off + r_ * jq2))) * E_DIM
                  + head * HDIM;
    float inv = 1.0f / lrun[j];
#pragma unroll
    for (int nt = 0; nt < 4; ++nt)
      att[orow + nt * 16 + ln] = o[nt][j] * inv;
  }
}

// ---------------------------------------------------------------------------
// cs[b, e] = sum over N of att[b, :, e]   (the x.sum(axis=(1,2)) term)
// ---------------------------------------------------------------------------
__global__ __launch_bounds__(256) void colsum_kernel(
    const float* __restrict__ att, float* __restrict__ cs)
{
  int idx = blockIdx.x * 256 + threadIdx.x;       // 0..1535 -> (b, e)
  int b = idx / E_DIM, col = idx % E_DIM;
  int p0 = blockIdx.y * 256;
  float acc = 0.0f;
  for (int p = 0; p < 256; ++p)
    acc += att[((size_t)b * SEQ_N + p0 + p) * E_DIM + col];
  atomicAdd(&cs[idx], acc);
}

// ---------------------------------------------------------------------------
// y = LN(att / (cs*3)) * w + b   -> bf16 for the final GEMM
// ---------------------------------------------------------------------------
__global__ __launch_bounds__(256) void ln_kernel(
    const float* __restrict__ att, const float* __restrict__ cs,
    const float* __restrict__ lnw, const float* __restrict__ lnb,
    __bf16* __restrict__ out)
{
  __shared__ float rsum[256], rsq[256];
  int row = blockIdx.x;
  int b   = row >> 13;                 // row / 8192
  int tid = threadIdx.x;
  float x[3]; float s = 0.0f, q = 0.0f;
#pragma unroll
  for (int i = 0; i < 3; ++i) {
    int e = i * 256 + tid;
    float v = att[(size_t)row * E_DIM + e] / (cs[b * E_DIM + e] * 3.0f);
    x[i] = v; s += v; q += v * v;
  }
  rsum[tid] = s; rsq[tid] = q;
  __syncthreads();
  for (int st = 128; st > 0; st >>= 1) {
    if (tid < st) { rsum[tid] += rsum[tid + st]; rsq[tid] += rsq[tid + st]; }
    __syncthreads();
  }
  float mu   = rsum[0] * (1.0f / E_DIM);
  float var  = rsq[0] * (1.0f / E_DIM) - mu * mu;
  float rstd = rsqrtf(var + 1e-5f);
#pragma unroll
  for (int i = 0; i < 3; ++i) {
    int e = i * 256 + tid;
    out[(size_t)row * E_DIM + e] = f2bf((x[i] - mu) * rstd * lnw[e] + lnb[e]);
  }
}

// ---------------------------------------------------------------------------
extern "C" void kernel_launch(void* const* d_in, const int* in_sizes, int n_in,
                              void* d_out, int out_size, void* d_ws, size_t ws_size,
                              hipStream_t stream)
{
  (void)in_sizes; (void)n_in; (void)out_size; (void)ws_size;
  const float* query = (const float*)d_in[0];
  const float* key_  = (const float*)d_in[1];
  const float* value = (const float*)d_in[2];
  const float* Wq = (const float*)d_in[3];
  const float* bq = (const float*)d_in[4];
  const float* Wk = (const float*)d_in[5];
  const float* bk = (const float*)d_in[6];
  const float* Wv = (const float*)d_in[7];
  const float* bv = (const float*)d_in[8];
  const float* Wo = (const float*)d_in[9];
  const float* bo = (const float*)d_in[10];
  const float* lnw = (const float*)d_in[11];
  const float* lnb = (const float*)d_in[12];

  const int M = 2 * SEQ_N;                       // 16384 rows
  char* ws = (char*)d_ws;
  auto take = [&](size_t bytes) {
    char* p = ws; ws += (bytes + 255) & ~(size_t)255; return p;
  };
  __bf16* qbuf  = (__bf16*)take((size_t)M * E_DIM * 2);
  __bf16* kbuf  = (__bf16*)take((size_t)M * E_DIM * 2);
  __bf16* vbuf  = (__bf16*)take((size_t)M * E_DIM * 2);
  float*  att   = (float*) take((size_t)M * E_DIM * 4);
  float*  cs    = (float*) take((size_t)2 * E_DIM * 4);
  __bf16* lnbuf = (__bf16*)take((size_t)M * E_DIM * 2);

  hipMemsetAsync(att, 0, (size_t)M * E_DIM * 4, stream);
  hipMemsetAsync(cs,  0, (size_t)2 * E_DIM * 4, stream);

  dim3 gg(M / 64, E_DIM / 128);
  gemm768_kernel<float, true><<<gg, 256, 0, stream>>>(query, Wq, bq, qbuf, M);
  gemm768_kernel<float, true><<<gg, 256, 0, stream>>>(key_,  Wk, bk, kbuf, M);
  gemm768_kernel<float, true><<<gg, 256, 0, stream>>>(value, Wv, bv, vbuf, M);

  dilated_attn_kernel<<<1792, 128, 0, stream>>>(qbuf, kbuf, vbuf, att);

  colsum_kernel<<<dim3(6, 32), 256, 0, stream>>>(att, cs);
  ln_kernel<<<M, 256, 0, stream>>>(att, cs, lnw, lnb, lnbuf);

  gemm768_kernel<__bf16, false><<<gg, 256, 0, stream>>>(lnbuf, Wo, bo, d_out, M);
}
